// MultiHeadedAttention_63264868270463
// MI455X (gfx1250) — compile-verified, matrix operand order checked by reading, not run on hardware
//
#include <hip/hip_runtime.h>
#include <hip/hip_bf16.h>

#define NH   16
#define DK   64
#define B_   2
#define S_   2048
#define DM   1024
#define MTOT (B_ * S_)   // 4096 token rows

typedef __bf16 bf16;
typedef __attribute__((ext_vector_type(16))) __bf16 v16bf;
typedef __attribute__((ext_vector_type(8)))  __bf16 v8bf;
typedef __attribute__((ext_vector_type(8)))  float  v8f;

// ---------------------------------------------------------------------------
// 16x32 bf16 operand fragment, row-major source, per documented CDNA5 layout:
//   lanes 0-15 : row = lane,    K = {0..7} U {16..23}
//   lanes 16-31: row = lane-16, K = {8..15} U {24..31}
// Works for both A and B operands (B operand == A-layout of B^T rows).
// ---------------------------------------------------------------------------
__device__ __forceinline__ v16bf frag_load(const bf16* __restrict__ base, int stride) {
    const int lane = threadIdx.x & 31;
    const int row  = lane & 15;
    const int kb   = (lane >> 4) << 3;    // 0 or 8
    const bf16* p  = base + row * stride + kb;
    v8bf lo = *(const v8bf*)(p);
    v8bf hi = *(const v8bf*)(p + 16);
    v16bf r;
#pragma unroll
    for (int i = 0; i < 8; ++i) { r[i] = lo[i]; r[i + 8] = hi[i]; }
    return r;
}

__device__ __forceinline__ v8f wmma_bf16(v16bf a, v16bf b, v8f c) {
    return __builtin_amdgcn_wmma_f32_16x16x32_bf16(
        /*neg_a=*/false, a, /*neg_b=*/false, b,
        /*c_mod=*/(short)0, c, /*reuse_a=*/false, /*reuse_b=*/false);
}

// ---------------------------------------------------------------------------
// f32 -> bf16 elementwise convert
// ---------------------------------------------------------------------------
__global__ void k_cvt(const float* __restrict__ src, bf16* __restrict__ dst, int n) {
    int i = blockIdx.x * blockDim.x + threadIdx.x;
    if (i < n) dst[i] = (bf16)src[i];
}

// W[k][n] (KxN f32) -> WT[n][k] (NxK bf16); coalesced along k on the write side
__global__ void k_transpose_cvt(const float* __restrict__ W, bf16* __restrict__ WT,
                                int K, int N) {
    int idx = blockIdx.x * blockDim.x + threadIdx.x;
    if (idx >= K * N) return;
    int k = idx % K;
    int n = idx / K;
    WT[n * K + k] = (bf16)W[k * N + n];
}

// ---------------------------------------------------------------------------
// Projection GEMM: out = A[MTOT x DM](bf16) @ W + bias, via BT = W^T (bf16).
// Block = 128 threads (4 waves); wave w owns one 16x16 N-tile; grid (M/16, N/64).
// mode 0: write [b, h, s, 64]   (Q, K)
// mode 1: write [b, h, 64, s]   (V transposed, for PV B operand)
// ---------------------------------------------------------------------------
__global__ void k_proj_gemm(const bf16* __restrict__ A, const bf16* __restrict__ BT,
                            const float* __restrict__ bias, bf16* __restrict__ dst,
                            int mode) {
    const int w    = threadIdx.x >> 5;
    const int lane = threadIdx.x & 31;
    const int m0   = blockIdx.x * 16;
    const int n0   = (blockIdx.y * 4 + w) * 16;

    const bf16* abase = A  + m0 * DM;
    const bf16* bbase = BT + n0 * DM;
    v8f acc = {};
#pragma unroll 4
    for (int kc = 0; kc < DM; kc += 32) {
        v16bf a = frag_load(abase + kc, DM);
        v16bf b = frag_load(bbase + kc, DM);
        acc = wmma_bf16(a, b, acc);
    }

    const int n    = n0 + (lane & 15);
    const int mh   = (lane >> 4) * 8;       // D layout: lanes>=16 hold rows 8..15
    const float bv = bias[n];
    const int h = n >> 6, d = n & 63;
#pragma unroll
    for (int r = 0; r < 8; ++r) {
        int m = m0 + mh + r;
        int bi = m >> 11;                   // / 2048
        int s  = m & 2047;
        float v = acc[r] + bv;
        if (mode == 0) dst[(((bi * NH + h) * S_) + s) * DK + d] = (bf16)v;
        else           dst[(((bi * NH + h) * DK) + d) * S_ + s] = (bf16)v;
    }
}

// ---------------------------------------------------------------------------
// Fused attention: one block (8 waves, 256 thr) = 16 query rows of one (b,h).
// LDS: sS[16][2048] f32 (128K) | sP[16][2048] bf16 (64K) | redA/redB | sO[2][16][64]
// Phase 1: S = Q K^T * 0.125 via WMMA -> LDS (16 k-tiles per wave)
// Phase 2: row softmax in LDS (16 thr/row); attn f32 -> d_out, bf16 -> sP
// Phase 3: O = P V via WMMA; k-range split across wave pairs, combined in LDS
// ---------------------------------------------------------------------------
#define OFF_SP (16 * S_ * 4)
#define OFF_RA (OFF_SP + 16 * S_ * 2)
#define OFF_RB (OFF_RA + 16 * 16 * 4)
#define OFF_SO (OFF_RB + 16 * 16 * 4)
#define SH_ATT (OFF_SO + 2 * 16 * 64 * 4)   // 206848 bytes < 320KB

__global__ void k_attention(const bf16* __restrict__ Qp, const bf16* __restrict__ Kp,
                            const bf16* __restrict__ VpT,
                            float* __restrict__ attn_out, bf16* __restrict__ Obuf) {
    extern __shared__ __align__(16) char smem[];
    float* sS   = (float*)smem;                 // 16*2048 f32 scores
    bf16*  sP   = (bf16*)(smem + OFF_SP);       // 16*2048 bf16 probs
    float* redA = (float*)(smem + OFF_RA);      // 16*16
    float* redB = (float*)(smem + OFF_RB);      // 16*16
    float* sO   = (float*)(smem + OFF_SO);      // 2*16*64 partial PV sums

    const int qt   = blockIdx.x;
    const int h    = blockIdx.y;
    const int bb   = blockIdx.z;
    const int tid  = threadIdx.x;
    const int w    = tid >> 5;
    const int lane = tid & 31;

    const bf16* qbase = Qp + ((bb * NH + h) * S_ + qt * 16) * DK;
    const bf16* kbase = Kp + ((bb * NH + h) * S_) * DK;

    // Q operand fragments (d = 0..63 -> two K=32 chunks), reused for all k tiles
    v16bf aq0 = frag_load(qbase, DK);
    v16bf aq1 = frag_load(qbase + 32, DK);

    // ---- Phase 1: scores -> LDS (wave w owns k-tiles [w*16, w*16+16)) ----
    for (int kt = w * 16; kt < w * 16 + 16; ++kt) {
        const bf16* kb_ = kbase + kt * 16 * DK;
        __builtin_prefetch(kb_ + 16 * DK, 0, 0);     // global_prefetch_b8: next tile
        v16bf b0 = frag_load(kb_, DK);
        v16bf b1 = frag_load(kb_ + 32, DK);
        v8f acc = {};
        acc = wmma_bf16(aq0, b0, acc);
        acc = wmma_bf16(aq1, b1, acc);
        const int n  = kt * 16 + (lane & 15);
        const int mh = (lane >> 4) * 8;
#pragma unroll
        for (int r = 0; r < 8; ++r)
            sS[(mh + r) * S_ + n] = acc[r] * 0.125f;   // 1/sqrt(64)
    }
    __syncthreads();

    // ---- Phase 2: softmax (16 threads per row) ----
    const int row = tid >> 4;
    const int t16 = tid & 15;
    float* srow = sS + row * S_;

    float lmax = -1e30f;
    for (int j = t16; j < S_; j += 16) lmax = fmaxf(lmax, srow[j]);
    redA[row * 16 + t16] = lmax;
    __syncthreads();
    float rmax = redA[row * 16];
#pragma unroll
    for (int i = 1; i < 16; ++i) rmax = fmaxf(rmax, redA[row * 16 + i]);

    float lsum = 0.f;
    for (int j = t16; j < S_; j += 16) {
        float e = __expf(srow[j] - rmax);
        srow[j] = e;
        lsum += e;
    }
    redB[row * 16 + t16] = lsum;   // distinct buffer: no race with redA readers
    __syncthreads();
    float rsum = redB[row * 16];
#pragma unroll
    for (int i = 1; i < 16; ++i) rsum += redB[row * 16 + i];
    const float inv = 1.0f / rsum;

    // attn layout: (h, b, q, k) reshaped to (h*b, q, k)
    float* arow = attn_out + ((h * B_ + bb) * S_ + qt * 16 + row) * S_;
    for (int j = t16; j < S_; j += 16) {
        float p = srow[j] * inv;
        arow[j] = p;
        sP[row * S_ + j] = (bf16)p;
    }
    __syncthreads();

    // ---- Phase 3: O = P @ V ----
    // wave pair (w, w+4): d-tile dt = w&3; k-half kh = w>>2 (each 1024 deep)
    const int dt = w & 3;
    const int kh = w >> 2;
    const bf16* vbase = VpT + ((bb * NH + h) * DK + dt * 16) * S_;
    v8f acc = {};
#pragma unroll 4
    for (int kc = kh * 1024; kc < kh * 1024 + 1024; kc += 32) {
        v16bf a = frag_load(sP + kc, S_);        // LDS operand
        v16bf b = frag_load(vbase + kc, S_);
        acc = wmma_bf16(a, b, acc);
    }
    const int n  = dt * 16 + (lane & 15);
    const int mh = (lane >> 4) * 8;
#pragma unroll
    for (int r = 0; r < 8; ++r)
        sO[(kh * 16 + mh + r) * 64 + n] = acc[r];
    __syncthreads();

    // combine the two k-half partials and store bf16 to Obuf [b, s, 1024]
    for (int idx = tid; idx < 16 * 64; idx += 256) {
        int r = idx >> 6, c = idx & 63;
        float vsum = sO[r * 64 + c] + sO[(16 + r) * 64 + c];
        Obuf[(bb * S_ + qt * 16 + r) * DM + h * DK + c] = (bf16)vsum;
    }
}

// ---------------------------------------------------------------------------
// Y = Obuf @ Wo + bo + residual, then LayerNorm. Block = 256 thr (8 waves),
// one block per 16 token rows; each wave does 8 N-tiles of 16.
// ---------------------------------------------------------------------------
__global__ void k_out_ln(const bf16* __restrict__ Obuf, const bf16* __restrict__ WoT,
                         const float* __restrict__ bo, const float* __restrict__ residual,
                         const float* __restrict__ ln_scale, const float* __restrict__ ln_bias,
                         float* __restrict__ y) {
    extern __shared__ __align__(16) char smem[];
    float* sX   = (float*)smem;                   // 16*1024 f32
    float* redA = (float*)(smem + 16 * DM * 4);   // 16*16
    float* redB = redA + 256;                     // 16*16

    const int m0   = blockIdx.x * 16;
    const int tid  = threadIdx.x;
    const int w    = tid >> 5;
    const int lane = tid & 31;

    const bf16* abase = Obuf + m0 * DM;
#pragma unroll
    for (int i = 0; i < 8; ++i) {
        const int n0 = (w * 8 + i) * 16;
        const bf16* bbase = WoT + n0 * DM;
        v8f acc = {};
#pragma unroll 4
        for (int kc = 0; kc < DM; kc += 32) {
            v16bf a = frag_load(abase + kc, DM);
            v16bf b = frag_load(bbase + kc, DM);
            acc = wmma_bf16(a, b, acc);
        }
        const int n    = n0 + (lane & 15);
        const int mh   = (lane >> 4) * 8;
        const float bv = bo[n];
#pragma unroll
        for (int r = 0; r < 8; ++r) {
            int m = m0 + mh + r;
            sX[(mh + r) * DM + n] = acc[r] + bv + residual[m * DM + n];
        }
    }
    __syncthreads();

    // LayerNorm: 16 threads per row
    const int row = tid >> 4;
    const int t16 = tid & 15;
    float* xr = sX + row * DM;
    float s1 = 0.f, s2 = 0.f;
    for (int j = t16; j < DM; j += 16) { float v = xr[j]; s1 += v; s2 += v * v; }
    redA[row * 16 + t16] = s1;
    redB[row * 16 + t16] = s2;
    __syncthreads();
    float m1 = 0.f, m2 = 0.f;
#pragma unroll
    for (int i = 0; i < 16; ++i) { m1 += redA[row * 16 + i]; m2 += redB[row * 16 + i]; }
    m1 *= (1.0f / DM);
    m2 = m2 * (1.0f / DM) - m1 * m1;                 // E[x^2] - mean^2
    const float rstd = rsqrtf(m2 + 1e-6f);
    const int gm = m0 + row;
    for (int j = t16; j < DM; j += 16)
        y[gm * DM + j] = (xr[j] - m1) * rstd * ln_scale[j] + ln_bias[j];
}

// ---------------------------------------------------------------------------
extern "C" void kernel_launch(void* const* d_in, const int* in_sizes, int n_in,
                              void* d_out, int out_size, void* d_ws, size_t ws_size,
                              hipStream_t stream) {
    (void)in_sizes; (void)n_in; (void)out_size; (void)ws_size;
    const float* q    = (const float*)d_in[0];
    const float* k    = (const float*)d_in[1];
    const float* v    = (const float*)d_in[2];
    const float* Wq   = (const float*)d_in[3];
    const float* bq   = (const float*)d_in[4];
    const float* Wk   = (const float*)d_in[5];
    const float* bk   = (const float*)d_in[6];
    const float* Wv   = (const float*)d_in[7];
    const float* bv   = (const float*)d_in[8];
    const float* Wo   = (const float*)d_in[9];
    const float* bo   = (const float*)d_in[10];
    const float* ln_s = (const float*)d_in[11];
    const float* ln_b = (const float*)d_in[12];

    float* y    = (float*)d_out;
    float* attn = y + (size_t)MTOT * DM;

    // Workspace layout (bf16 buffers), total 64 MB
    const size_t ACT = (size_t)MTOT * DM;   // 4M elems
    const size_t WEL = (size_t)DM * DM;     // 1M elems
    bf16* qbf  = (bf16*)d_ws;
    bf16* kbf  = qbf + ACT;
    bf16* vbf  = kbf + ACT;
    bf16* WqT  = vbf + ACT;
    bf16* WkT  = WqT + WEL;
    bf16* WvT  = WkT + WEL;
    bf16* WoT  = WvT + WEL;
    bf16* Qp   = WoT + WEL;
    bf16* Kp   = Qp  + ACT;
    bf16* VpT  = Kp  + ACT;
    bf16* Obuf = VpT + ACT;

    const int nA = (int)ACT, nW = (int)WEL;
    k_cvt<<<(nA + 255) / 256, 256, 0, stream>>>(q, qbf, nA);
    k_cvt<<<(nA + 255) / 256, 256, 0, stream>>>(k, kbf, nA);
    k_cvt<<<(nA + 255) / 256, 256, 0, stream>>>(v, vbf, nA);
    k_transpose_cvt<<<(nW + 255) / 256, 256, 0, stream>>>(Wq, WqT, DM, DM);
    k_transpose_cvt<<<(nW + 255) / 256, 256, 0, stream>>>(Wk, WkT, DM, DM);
    k_transpose_cvt<<<(nW + 255) / 256, 256, 0, stream>>>(Wv, WvT, DM, DM);
    k_transpose_cvt<<<(nW + 255) / 256, 256, 0, stream>>>(Wo, WoT, DM, DM);

    dim3 gp(MTOT / 16, DM / 64);            // 256 x 16
    k_proj_gemm<<<gp, 128, 0, stream>>>(qbf, WqT, bq, Qp, 0);
    k_proj_gemm<<<gp, 128, 0, stream>>>(kbf, WkT, bk, Kp, 0);
    k_proj_gemm<<<gp, 128, 0, stream>>>(vbf, WvT, bv, VpT, 1);

    dim3 ga(S_ / 16, NH, B_);               // 128 x 16 x 2
    k_attention<<<ga, 256, (size_t)SH_ATT, stream>>>(Qp, Kp, VpT, attn, Obuf);

    size_t shF = (size_t)16 * DM * 4 + 2048;
    k_out_ln<<<MTOT / 16, 256, shF, stream>>>(Obuf, WoT, bo, q, ln_s, ln_b, y);
}
